// SoftmaxTopKMaxPooling_Prefill_Fused_15547781612184
// MI455X (gfx1250) — matrix operand order repaired and checked by reading, not compiled, not run on hardware
//
#include <hip/hip_runtime.h>
#include <hip/hip_bf16.h>
#include <math.h>

// CDNA5 / gfx1250 wave32 WMMA types
typedef __attribute__((ext_vector_type(16))) __bf16 v16bf;
typedef __attribute__((ext_vector_type(8)))  __bf16 v8bf;
typedef __attribute__((ext_vector_type(8)))  float  v8f;

#define LQ   4096   // query length
#define HN   4      // kv heads
#define GN   16     // GQA group (== WMMA M tile!)
#define DN   128    // head dim
#define SN   64     // number of key blocks (L / BLOCK_SIZE)
#define WIN  256    // WINDOW_SIZE

// One wave per (b, h, l): computes the full 16(g) x 64(s) score tile with
// 16x v_wmma_f32_16x16x32_bf16 and reduces masked LSE over s in-register.
__global__ __launch_bounds__(256) void nsa_cmp_lse_kernel(
    const __hip_bfloat16* __restrict__ Qh,
    const __hip_bfloat16* __restrict__ Kh,
    const int*            __restrict__ seq_lens,
    float*                __restrict__ out)
{
    const __bf16* __restrict__ Q = (const __bf16*)Qh;
    const __bf16* __restrict__ K = (const __bf16*)Kh;

    const int lane = threadIdx.x & 31;
    const int wid  = threadIdx.x >> 5;
    const int gw   = blockIdx.x * 8 + wid;      // global wave id
    const int l = gw % LQ;
    const int h = (gw / LQ) % HN;
    const int b = gw / (LQ * HN);

    const int n   = lane & 15;   // A-row (g) for loads, C/D column (s%16)
    const int hlf = lane >> 4;   // lane half

    // Q[b, l, h, g=n, d]  (d contiguous)
    const __bf16* qrow = Q + ((size_t)((b * LQ + l) * HN + h) * GN + n) * DN;
    // K[b, s, h, d]: base for (b, h); per-s stride = HN*DN elements
    const __bf16* kbas = K + ((size_t)b * SN * HN + h) * DN;

    v8f c[4] = {};   // 4 N-tiles of 16x16 f32 accumulators

    #pragma unroll
    for (int kk = 0; kk < 4; ++kk) {           // K dimension: 4 x 32
        // ---- A fragment: row m = lane%16, K-values hlf*8+{0..7} and 16+hlf*8+{0..7}
        union { v16bf v; v8bf p[2]; } ua;
        const int dA = kk * 32 + hlf * 8;
        ua.p[0] = *(const v8bf*)(qrow + dA);
        ua.p[1] = *(const v8bf*)(qrow + dA + 16);
        const v16bf afrag = ua.v;

        #pragma unroll
        for (int nt = 0; nt < 4; ++nt) {       // 4 column tiles of s
            // ---- B fragment: column n, K-values hlf*8+{0..7} and 16+hlf*8+{0..7}
            const int s = nt * 16 + n;
            const __bf16* kp = kbas + (size_t)s * (HN * DN) + kk * 32 + hlf * 8;
            union { v16bf v; v8bf p[2]; } ub;
            ub.p[0] = *(const v8bf*)(kp);
            ub.p[1] = *(const v8bf*)(kp + 16);
            c[nt] = __builtin_amdgcn_wmma_f32_16x16x32_bf16(
                        false, afrag, false, ub.v, (short)0, c[nt], false, false);
        }
    }

    // ---- wave-uniform column limit: s valid iff s < min((l-255)//64, slen//64)
    const int slen = seq_lens[b];
    int t   = l - (WIN - 1);
    int thr = (t >= 0) ? (t >> 6) : 0;          // floor, clamped (neg -> 0 valid cols)
    const int limit = min(thr, slen >> 6);
    const float scale = 0.08838834764831845f;   // 1/sqrt(128)

    // ---- masked LSE over s for each of the 8 row-slots this lane holds
    float lsev[8];
    #pragma unroll
    for (int i = 0; i < 8; ++i) {
        float v0 = ((n     ) < limit) ? c[0][i] * scale : -1e30f;
        float v1 = ((n + 16) < limit) ? c[1][i] * scale : -1e30f;
        float v2 = ((n + 32) < limit) ? c[2][i] * scale : -1e30f;
        float v3 = ((n + 48) < limit) ? c[3][i] * scale : -1e30f;

        float m = fmaxf(fmaxf(v0, v1), fmaxf(v2, v3));
        m = fmaxf(m, __shfl_xor(m, 1, 32));
        m = fmaxf(m, __shfl_xor(m, 2, 32));
        m = fmaxf(m, __shfl_xor(m, 4, 32));
        m = fmaxf(m, __shfl_xor(m, 8, 32));     // row max across 16 lanes of half

        const float msafe = (m < -0.5e30f) ? 0.0f : m;
        float s4 = __expf(v0 - msafe) + __expf(v1 - msafe)
                 + __expf(v2 - msafe) + __expf(v3 - msafe);
        s4 += __shfl_xor(s4, 1, 32);
        s4 += __shfl_xor(s4, 2, 32);
        s4 += __shfl_xor(s4, 4, 32);
        s4 += __shfl_xor(s4, 8, 32);            // row sum across 16 lanes of half

        lsev[i] = (s4 > 0.0f) ? (msafe + __logf(s4)) : -INFINITY;
    }

    // ---- lanes n==0 (lane 0 -> g=0..7, lane 16 -> g=8..15) write 8 f32 each
    if (n == 0) {
        const bool qv = (l < slen);
        float* op = out + ((size_t)((b * LQ + l) * HN + h) * GN) + hlf * 8;
        float r[8];
        #pragma unroll
        for (int i = 0; i < 8; ++i) r[i] = qv ? lsev[i] : -INFINITY;
        *(float4*)(op)     = make_float4(r[0], r[1], r[2], r[3]);
        *(float4*)(op + 4) = make_float4(r[4], r[5], r[6], r[7]);
    }
}

extern "C" void kernel_launch(void* const* d_in, const int* in_sizes, int n_in,
                              void* d_out, int out_size, void* d_ws, size_t ws_size,
                              hipStream_t stream) {
    const __hip_bfloat16* Q  = (const __hip_bfloat16*)d_in[0];
    const __hip_bfloat16* Kp = (const __hip_bfloat16*)d_in[1];
    const int* seq_lens      = (const int*)d_in[2];
    float* out               = (float*)d_out;

    const int B = in_sizes[0] / (LQ * HN * GN * DN);   // 2 for the reference setup
    const int total_waves = B * HN * LQ;               // one wave per (b,h,l)
    const int blocks = total_waves / 8;                // 8 waves (256 thr) per block

    nsa_cmp_lse_kernel<<<blocks, 256, 0, stream>>>(Q, Kp, seq_lens, out);
}
